// RosaQKVLayer_77094662963960
// MI455X (gfx1250) — matrix operand order repaired and checked by reading, not compiled, not run on hardware
//
#include <hip/hip_runtime.h>

typedef _Float16 v16h __attribute__((ext_vector_type(16)));
typedef _Float16 h8   __attribute__((ext_vector_type(8)));
typedef _Float16 h4   __attribute__((ext_vector_type(4)));
typedef float    v8f  __attribute__((ext_vector_type(8)));
typedef unsigned int  u32x4 __attribute__((ext_vector_type(4)));
typedef int           i32x4 __attribute__((ext_vector_type(4)));
typedef int           i32x8 __attribute__((ext_vector_type(8)));
typedef unsigned long long u64;

#define B_   4
#define T_   4096
#define D_   1024
#define H_   16
#define M_   (B_*T_)              // 16384 rows
#define BROW_ 2064                // padded LDS row pitch in bytes (1032 halfs)
#define PANEL_BYTES (64 * BROW_)  // 132096 B per workgroup

#if defined(__has_builtin)
#if __has_builtin(__builtin_amdgcn_tensor_load_to_lds)
#define HAVE_TDM 1
#endif
#endif

// ---------------- fp32 -> fp16 staging (4 elems / thread) ----------------
__global__ void cvt_f32_f16(const float* __restrict__ s,
                            _Float16* __restrict__ d, int n4) {
  int i = blockIdx.x * blockDim.x + threadIdx.x;
  if (i >= n4) return;
  const float4 f = reinterpret_cast<const float4*>(s)[i];
  h4 o;
  o[0] = (_Float16)f.x; o[1] = (_Float16)f.y;
  o[2] = (_Float16)f.z; o[3] = (_Float16)f.w;
  reinterpret_cast<h4*>(d)[i] = o;
}

// ------------- WMMA GEMM + sign-bit packing --------------------------------
// block = 128 threads (4 waves). Each block: 4 M-tiles x one 64-col N group.
// B panel (64 rows x 1024 halfs) staged into LDS once via TDM, padded rows.
// grid.x = M_/64 (256), grid.y = 24 (4 q-groups, 4 k-groups, 16 v-groups)
__global__ __launch_bounds__(128) void gemm_signpack(
    const _Float16* __restrict__ xh,   // (16384,1024) row-major f16
    const _Float16* __restrict__ wq,   // (256,1024)
    const _Float16* __restrict__ wk,   // (256,1024)
    const _Float16* __restrict__ wv,   // (1024,1024)
    u64* __restrict__ qc, u64* __restrict__ kc, u64* __restrict__ vc) {
  extern __shared__ char smem[];                 // PANEL_BYTES
  const int tid  = threadIdx.x;
  const int lane = tid & 31;
  const int wave = tid >> 5;
  const int g    = blockIdx.y;
  const _Float16* w; int n0, mat;
  if (g < 4)       { w = wq; n0 = g * 64;       mat = 0; }
  else if (g < 8)  { w = wk; n0 = (g - 4) * 64; mat = 1; }
  else             { w = wv; n0 = (g - 8) * 64; mat = 2; }

  const _Float16* wpanel = w + (size_t)n0 * D_;  // 64 rows x 1024 halfs

#ifdef HAVE_TDM
  if (wave == 0) {
    // --- Tensor DMA descriptor: 64 iterated rows of 2048B, LDS pitch 2064B ---
    const u64 ga = (u64)(size_t)wpanel;
    const unsigned lds_off = (unsigned)(size_t)smem;   // LDS aperture low bits
    u32x4 g0;
    g0[0] = 1u;                                   // count=1, user descriptor
    g0[1] = lds_off;                              // lds_addr
    g0[2] = (unsigned)(ga & 0xffffffffu);         // global_addr[31:0]
    g0[3] = (unsigned)((ga >> 32) & 0x01ffffffu) | 0x80000000u; // ga[56:32] | type=2
    i32x8 g1;
    g1[0] = (int)0x00090000;   // data_size=2B (bit16), iterate_enable (bit19)
    g1[1] = (int)(1024u << 16);// tensor_dim0[15:0]=1024 in bits 63:48
    g1[2] = (int)(64u << 16);  // tensor_dim1[15:0]=64 in bits 95:80
    g1[3] = (int)(1024u << 16);// tile_dim0=1024 in bits 127:112
    g1[4] = 1;                 // tile_dim1=1 (one row per iteration)
    g1[5] = 1024;              // tensor_dim0_stride
    g1[6] = 0;
    g1[7] = 0;
    i32x4 g2;
    g2[0] = 0;                 // tensor_dim2 (unused, iterate mode)
    g2[1] = 1032;              // lds_addr_increment (halfs) -> 2064B pitch
    g2[2] = 1024;              // global_addr_increment (halfs)
    g2[3] = (int)(63u << 16);  // iterate_count=63 -> 64 iterations
    i32x4 g3; g3[0] = 0; g3[1] = 0; g3[2] = 0; g3[3] = 0;
#if __clang_major__ >= 23
    i32x8 gz; gz[0]=0;gz[1]=0;gz[2]=0;gz[3]=0;gz[4]=0;gz[5]=0;gz[6]=0;gz[7]=0;
    __builtin_amdgcn_tensor_load_to_lds(g0, g1, g2, g3, gz, 0);
#else
    __builtin_amdgcn_tensor_load_to_lds(g0, g1, g2, g3, 0);
#endif
    __builtin_amdgcn_s_wait_tensorcnt(0);
  }
#else
  // Fallback: cooperative copy with the same padded pitch.
  for (int c = tid; c < 64 * 128; c += 128) {    // 16B chunks: 64 rows x 128
    const int r = c >> 7, o = (c & 127) * 16;
    *reinterpret_cast<h8*>(smem + (size_t)r * BROW_ + o) =
        *reinterpret_cast<const h8*>(
            reinterpret_cast<const char*>(wpanel) + (size_t)r * 2048 + o);
  }
#endif
  __syncthreads();

  const int m0  = (blockIdx.x * 4 + wave) * 16;
  const int row = lane & 15;
  const int hi  = lane >> 4;

  // A lane layout (16-bit 16x32): halfs0-7 = K 8*hi+0..7, halfs8-15 = K 16+8*hi+0..7
  const _Float16* a0 = xh + (size_t)(m0 + row) * D_ + 8 * hi;
  // B lane layout (16-bit 32x16): halfs0-7 = K 8*hi+0..7, halfs8-15 = K 16+8*hi+0..7, col = lane&15
  const char* bl0 = smem + (size_t)row * BROW_ + 16 * hi;

  v8f acc0 = {}, acc1 = {}, acc2 = {}, acc3 = {};
  union AU { v16h v; h8 h[2]; };
  AU acur, anxt;
  acur.h[0] = *reinterpret_cast<const h8*>(a0);
  acur.h[1] = *reinterpret_cast<const h8*>(a0 + 16);

  for (int k0 = 0; k0 < D_; k0 += 32) {
    const int kn = k0 + 32;
    if (kn < D_) {                       // prefetch next A fragment (regs)
      anxt.h[0] = *reinterpret_cast<const h8*>(a0 + kn);
      anxt.h[1] = *reinterpret_cast<const h8*>(a0 + kn + 16);
    }
    union BU { v16h v; h8 h[2]; } b0u, b1u, b2u, b3u;
#define BLOAD(J, BU_)                                                         \
    BU_.h[0] = *reinterpret_cast<const h8*>(bl0 + (size_t)(J) * 16 * BROW_ + 2 * k0); \
    BU_.h[1] = *reinterpret_cast<const h8*>(bl0 + (size_t)(J) * 16 * BROW_ + 2 * k0 + 32);
    BLOAD(0, b0u) BLOAD(1, b1u) BLOAD(2, b2u) BLOAD(3, b3u)
#undef BLOAD
    acc0 = __builtin_amdgcn_wmma_f32_16x16x32_f16(false, acur.v, false, b0u.v,
                                                  (short)0, acc0, false, false);
    acc1 = __builtin_amdgcn_wmma_f32_16x16x32_f16(false, acur.v, false, b1u.v,
                                                  (short)0, acc1, false, false);
    acc2 = __builtin_amdgcn_wmma_f32_16x16x32_f16(false, acur.v, false, b2u.v,
                                                  (short)0, acc2, false, false);
    acc3 = __builtin_amdgcn_wmma_f32_16x16x32_f16(false, acur.v, false, b3u.v,
                                                  (short)0, acc3, false, false);
    if (kn < D_) acur = anxt;
  }

  // C layout: VGPR r = (M=r, N=lane) lanes 0-15, (M=r+8, N=lane-16) lanes 16-31
  unsigned msk0[8], msk1[8], msk2[8], msk3[8];
#pragma unroll
  for (int r = 0; r < 8; ++r) {
    msk0[r] = __builtin_amdgcn_ballot_w32(acc0[r] > 0.0f);
    msk1[r] = __builtin_amdgcn_ballot_w32(acc1[r] > 0.0f);
    msk2[r] = __builtin_amdgcn_ballot_w32(acc2[r] > 0.0f);
    msk3[r] = __builtin_amdgcn_ballot_w32(acc3[r] > 0.0f);
  }

#pragma unroll
  for (int r = 0; r < 8; ++r) {
    if (lane == r || lane == 16 + r) {
      const bool hiRow = (lane != r);
      const int  rl = r + (hiRow ? 8 : 0);
      const int  m  = m0 + rl;
      const int  bb = m >> 12;
      const int  t  = m & (T_ - 1);
      const unsigned c0 = hiRow ? (msk0[r] >> 16) : (msk0[r] & 0xffffu);
      const unsigned c1 = hiRow ? (msk1[r] >> 16) : (msk1[r] & 0xffffu);
      const unsigned c2 = hiRow ? (msk2[r] >> 16) : (msk2[r] & 0xffffu);
      const unsigned c3 = hiRow ? (msk3[r] >> 16) : (msk3[r] & 0xffffu);
      if (mat == 2) {
        const int h = n0 >> 6;
        const u64 code = (u64)c0 | ((u64)c1 << 16) | ((u64)c2 << 32) | ((u64)c3 << 48);
        vc[(size_t)(bb * H_ + h) * T_ + t] = code;
      } else {
        u64* dst = (mat == 0) ? qc : kc;
        const int h0 = n0 >> 4;
        dst[(size_t)(bb * H_ + h0 + 0) * T_ + t] = c0;
        dst[(size_t)(bb * H_ + h0 + 1) * T_ + t] = c1;
        dst[(size_t)(bb * H_ + h0 + 2) * T_ + t] = c2;
        dst[(size_t)(bb * H_ + h0 + 3) * T_ + t] = c3;
      }
    }
  }
}

// --------- causal exact-match scan: 1 wave per (b,h), LDS u32 table ---------
__global__ __launch_bounds__(32) void rosa_scan(
    const u64* __restrict__ qc, const u64* __restrict__ kc,
    const u64* __restrict__ vc, u64* __restrict__ y) {
  extern __shared__ unsigned int table[];        // 2^16 entries = 256 KB
  const int lane = threadIdx.x;
  const int s = blockIdx.x;                      // 0..63 sequence id
  uint4* t4 = reinterpret_cast<uint4*>(table);
  for (int i = lane; i < (1 << 16) / 4; i += 32) t4[i] = make_uint4(0u, 0u, 0u, 0u);
  __syncthreads();

  const u64* q = qc + (size_t)s * T_;
  const u64* k = kc + (size_t)s * T_;
  const u64* v = vc + (size_t)s * T_;
  u64* yo = y + (size_t)s * T_;

  for (int base = 0; base < T_; base += 32) {
    const int t = base + lane;
    const unsigned myq = (unsigned)q[t];
    const unsigned myk = (unsigned)k[t];
    const unsigned tv = table[myq];              // latest j+1 with j < base
    int best = -1;                               // intra-chunk: latest j < lane
    for (int j = 0; j < 32; ++j) {
      const unsigned bk = __shfl(myk, j, 32);
      if (j < lane && bk == myq) best = j;
    }
    const u64 idx = (best >= 0) ? (u64)(base + best + 1) : (u64)tv;
    yo[t] = idx ? v[idx - 1] : 0ull;
    __syncthreads();
    atomicMax(&table[myk], (unsigned)(t + 1));   // ds_max_u32: latest t wins
    __syncthreads();
  }
}

// ------------------- bitmask -> emb select (streaming) -------------------
__global__ void expand_out(const u64* __restrict__ y,
                           const float* __restrict__ e0,
                           const float* __restrict__ e1,
                           float* __restrict__ out) {
  const int i = blockIdx.x * blockDim.x + threadIdx.x;
  const int d = i & (D_ - 1);
  const int m = i >> 10;
  const int b = m >> 12;
  const int t = m & (T_ - 1);
  const int h = d >> 6;
  const int j = d & 63;
  const u64 code = y[(size_t)(b * H_ + h) * T_ + t];
  out[i] = ((code >> j) & 1ull) ? e1[d] : e0[d];
}

extern "C" void kernel_launch(void* const* d_in, const int* in_sizes, int n_in,
                              void* d_out, int out_size, void* d_ws, size_t ws_size,
                              hipStream_t stream) {
  (void)in_sizes; (void)n_in; (void)out_size; (void)ws_size;
  const float* x  = (const float*)d_in[0];
  const float* Wq = (const float*)d_in[1];
  const float* Wk = (const float*)d_in[2];
  const float* Wv = (const float*)d_in[3];
  const float* e0 = (const float*)d_in[4];
  const float* e1 = (const float*)d_in[5];
  float* out = (float*)d_out;

  char* ws = (char*)d_ws;                       // ~43 MB used
  _Float16* xh  = (_Float16*)(ws);              // 16384*1024*2 = 33554432
  _Float16* wqh = (_Float16*)(ws + 33554432);   // 524288
  _Float16* wkh = (_Float16*)(ws + 34078720);   // 524288
  _Float16* wvh = (_Float16*)(ws + 34603008);   // 2097152
  u64* qc = (u64*)(ws + 36700160);              // 64*4096*8 = 2097152
  u64* kc = (u64*)(ws + 38797312);
  u64* vc = (u64*)(ws + 40894464);
  u64* yy = (u64*)(ws + 42991616);

  cvt_f32_f16<<<dim3((M_ * D_ / 4) / 256), 256, 0, stream>>>(x,  xh,  M_ * D_ / 4);
  cvt_f32_f16<<<dim3((256  * D_ / 4) / 256), 256, 0, stream>>>(Wq, wqh, 256  * D_ / 4);
  cvt_f32_f16<<<dim3((256  * D_ / 4) / 256), 256, 0, stream>>>(Wk, wkh, 256  * D_ / 4);
  cvt_f32_f16<<<dim3((1024 * D_ / 4) / 256), 256, 0, stream>>>(Wv, wvh, 1024 * D_ / 4);

  gemm_signpack<<<dim3(M_ / 64, 24), 128, PANEL_BYTES, stream>>>(
      xh, wqh, wkh, wvh, qc, kc, vc);

  rosa_scan<<<64, 32, (1 << 16) * sizeof(unsigned int), stream>>>(qc, kc, vc, yy);

  expand_out<<<(M_ * D_) / 256, 256, 0, stream>>>(yy, e0, e1, out);
}
